// KTBM_pers_nomat_37323265802341
// MI455X (gfx1250) — matrix operand order repaired and checked by reading, not compiled, not run on hardware
//
#include <hip/hip_runtime.h>

// KTBM recurrent knowledge-tracing model for gfx1250 (MI455X).
// 8 workgroups x 16 batch rows; persistent time loop; 3 fused f16 WMMA GEMMs
// per step (v_wmma_f32_16x16x32_f16). Loop-interchanged GEMMs: one A-fragment
// LDS load per K-chunk feeds 8 back-to-back WMMAs (accumulators held in
// registers). Weights pre-swizzled into B-fragment lane layout in workspace;
// DKVMN memory M streamed through L2 via workspace; fast rcp-based
// activations (no IEEE division sequences).

typedef __attribute__((ext_vector_type(16))) _Float16 v16h;
typedef __attribute__((ext_vector_type(8)))  float    v8f;

#define TT     100
#define NSTEP  98
#define KPAD   576          // LDS A-buffer row stride (halves)
// GEMM shapes
#define KT1 18              // K=576
#define NT1 32              // N=512
#define KT2 14              // K=448
#define NT2 32              // N=512
#define KT3 10              // K=320
#define NT3 8               // N=128
#define FRAG_HALVES 512     // 32 lanes * 16 halves

// ---- workspace layout (bytes) ----
#define WS_W1   0u
#define WS_W2   589824u                  // W1: 576 frags * 512 h * 2B
#define WS_W3   1048576u                 // W2: 448 frags * 512 h * 2B
#define WS_M    1130496u                 // W3: 80 frags  * 512 h * 2B
#define WS_BQ   3227648u                 // M: 128*32*128*4
#define WS_BL   3229696u                 // biasQ: 512 f32
#define WS_B2   3231744u                 // biasL: 512 f32
#define WS_BO1  3233792u                 // bias2: 512 f32; o1b: 128 f32

struct P {
  const int   *q_data, *a_data, *l_data, *d_data, *s_data;
  const float *init_M;
  const float *q_emb, *l_emb, *a_emb, *d_emb, *s_emb, *qb_emb, *lb_emb, *q_corr, *l_corr;
  const float *W_EQ, *b_EQ, *W_EL, *b_EL, *W_Ebh, *W_Estu, *b_Estu;
  const float *W_DQ, *b_DQ, *W_DL, *b_DL, *W_Dbh, *W_Dstu;
  const float *W_tr, *b_tr, *W_o1, *b_o1, *W_o2, *b_o2;
  const float *W_mQ, *b_mQ, *W_mL, *b_mL, *W_pref, *b_pref;
  const float *Wg[4], *bg[4], *Wgk[4], *bgk[4], *Wgh[4], *Wgs[4], *bgs[4];
  const float *w_k, *b_k;
};

__device__ __forceinline__ float rcp_f(float x) { return __builtin_amdgcn_rcpf(x); }
__device__ __forceinline__ float sigm(float x) { return rcp_f(1.0f + __expf(-x)); }
__device__ __forceinline__ float tanh_f(float x) {
  float a = fabsf(x);
  float e = __expf(-2.0f * a);
  float t = (1.0f - e) * rcp_f(1.0f + e);
  return x < 0.0f ? -t : t;
}

__device__ __forceinline__ v8f wmma_f16(v16h a, v16h b, v8f c) {
  return __builtin_amdgcn_wmma_f32_16x16x32_f16(false, a, false, b, (short)0, c, false, false);
}

// A-fragment (16x32 f16, M=batch-rows) from LDS row-major [16][KPAD].
// Per ISA: lanes 0-15 hold M=0-15 / K(0..7,16..23); lanes 16-31 same M, K+8.
__device__ __forceinline__ v16h load_afrag(const _Float16* uA, int kbase) {
  const int lane = threadIdx.x & 31;
  const int m = lane & 15, kh = lane >> 4;
  v16h a;
#pragma unroll
  for (int v = 0; v < 8; ++v) {
    int k0 = kbase + ((v < 4) ? 2 * v : 16 + 2 * (v - 4)) + 8 * kh;
    a[2 * v]     = uA[m * KPAD + k0];
    a[2 * v + 1] = uA[m * KPAD + k0 + 1];
  }
  return a;
}

// B-fragment: pre-swizzled so each lane reads 16 contiguous halves (32B).
__device__ __forceinline__ v16h load_bfrag(const _Float16* W, int frag) {
  const v16h* pv = (const v16h*)(W + (size_t)frag * FRAG_HALVES);
  return pv[threadIdx.x & 31];
}

// ---------- big stacked weight matrices (element mapping) ----------
// W1: K=576 rows [qa*is | l*ns | h | se | qb*is | lb*ns | d*is | d*ns | d | d1]
//     N=512 cols [erase | add | x | trans]
__device__ float w1_val(const P& p, int k, int n) {
  if (n < 128) {                      // erase
    if (k < 128) return p.W_EQ[k * 128 + n];
    if (k < 192) return p.W_EL[(k - 128) * 128 + n];
    if (k < 320) return p.W_Ebh[(k - 192) * 128 + n];
    if (k < 384) return p.W_Estu[(k - 320) * 128 + n];
    return 0.0f;
  } else if (n < 256) {               // add
    int c = n - 128;
    if (k < 128) return p.W_DQ[k * 128 + c];
    if (k < 192) return p.W_DL[(k - 128) * 128 + c];
    if (k < 320) return p.W_Dbh[(k - 192) * 128 + c];
    if (k < 384) return p.W_Dstu[(k - 320) * 128 + c];
    return 0.0f;
  } else if (n < 384) {               // behavior map x
    int c = n - 256;
    if (k >= 384 && k < 448) return p.W_mQ[(k - 384) * 128 + c];
    if (k >= 448 && k < 512) return p.W_mL[(k - 448) * 128 + c];
    if (k >= 512 && k < 528) return p.W_mQ[(64 + (k - 512)) * 128 + c];
    if (k >= 528 && k < 544) return p.W_mL[(64 + (k - 528)) * 128 + c];
    return 0.0f;
  } else {                            // trans
    int c = n - 384;
    if (k >= 544 && k < 576) return p.W_tr[(k - 544) * 128 + c];
    return 0.0f;
  }
}

// W2: K=448 rows [x | k | h | se], N=512 cols [i | g | f | o]
__device__ float w2_val(const P& p, int k, int n) {
  int g = n >> 7, c = n & 127;
  if (k < 128) return p.Wg[g][k * 128 + c];
  if (k < 256) return p.Wgk[g][(k - 128) * 128 + c];
  if (k < 384) return p.Wgh[g][(k - 256) * 128 + c];
  return p.Wgs[g][(k - 384) * 128 + c];
}

// ---------------- prep kernels ----------------
__global__ void prep_weights(P p, _Float16* W1, _Float16* W2, _Float16* W3) {
  const int F1 = KT1 * NT1, F2 = KT2 * NT2, F3 = KT3 * NT3;
  int idx = blockIdx.x * blockDim.x + threadIdx.x;
  int total = (F1 + F2 + F3) * FRAG_HALVES;
  if (idx >= total) return;
  int f = idx >> 9, w = idx & 511;
  int lane = w >> 4, h = w & 15;
  int kh = lane >> 4, nl = lane & 15;
  int kofs = ((h < 8) ? h : 8 + h) + 8 * kh;
  float val;
  _Float16* dst;
  if (f < F1) {
    int kt = f / NT1, nt = f % NT1;
    val = w1_val(p, kt * 32 + kofs, nt * 16 + nl);
    dst = W1 + idx;
  } else if (f < F1 + F2) {
    int fl = f - F1;
    int kt = fl / NT2, nt = fl % NT2;
    val = w2_val(p, kt * 32 + kofs, nt * 16 + nl);
    dst = W2 + (idx - F1 * FRAG_HALVES);
  } else {
    int fl = f - F1 - F2;
    int kt = fl / NT3, nt = fl % NT3;
    int k = kt * 32 + kofs, n = nt * 16 + nl;
    val = (k < 320) ? p.W_o1[k * 128 + n] : 0.0f;
    dst = W3 + (idx - (F1 + F2) * FRAG_HALVES);
  }
  *dst = (_Float16)val;
}

__global__ void prep_M(const float* src, float* dst) {
  int idx = blockIdx.x * blockDim.x + threadIdx.x;
  if (idx < 128 * 32 * 128) dst[idx] = src[idx];
}

// combined bias tables: biasQ/biasL (GEMM1, is_q select), bias2 (gates), o1b
__global__ void prep_bias(P p, float* bq, float* bl, float* b2, float* bo1) {
  int n = blockIdx.x * blockDim.x + threadIdx.x;
  if (n >= 512) return;
  int blk = n >> 7, c = n & 127;
  float vq, vl;
  if (blk == 0)      { vq = p.b_EQ[c] + p.b_Estu[c]; vl = p.b_EL[c] + p.b_Estu[c]; }
  else if (blk == 1) { vq = p.b_DQ[c]; vl = p.b_DL[c]; }
  else if (blk == 2) { vq = p.b_mQ[c]; vl = p.b_mL[c]; }
  else               { vq = p.b_tr[c]; vl = p.b_tr[c]; }
  bq[n] = vq;
  bl[n] = vl;
  b2[n] = p.bg[blk][c] + p.bgk[blk][c] + p.bgs[blk][c];
  if (n < 128) bo1[n] = p.b_o1[n];
}

// ---------------- main persistent kernel ----------------
__launch_bounds__(128)
__global__ void ktbm_main(P p, const _Float16* W1, const _Float16* W2,
                          const _Float16* W3, float* Mws,
                          const float* bqW, const float* blW, const float* b2W,
                          const float* bo1W, float* out) {
  __shared__ _Float16 uA[16 * KPAD];      // GEMM A staging (f16)
  __shared__ _Float16 outS[16 * 512];     // GEMM outputs (post-activation, f16)
  __shared__ float    hS[16 * 128];       // behavior hidden state
  __shared__ float    mS[16 * 128];       // LSTM cell state
  __shared__ _Float16 seS[16 * 64];       // student embedding (static)
  __shared__ _Float16 qnS[16 * 64];       // next-question embedding
  __shared__ _Float16 rdS[16 * 128];      // memory read (next-q correlation)
  __shared__ float    wcur[16 * 32];      // current correlation weights
  __shared__ float    wnxt[16 * 32];      // next-q correlation weights
  __shared__ float    wkS[32];            // w_k cached
  __shared__ float    wo2S[128];          // W_o2 cached
  __shared__ float    wprS[128];          // W_pref cached
  __shared__ float    isf[16];            // is_q as float mask
  __shared__ int      isq[16];
  __shared__ int      idxs[16][6];        // qi, ai, li, di, di0, qnext

  const int tid  = threadIdx.x;
  const int wave = tid >> 5;
  const int lane = tid & 31;
  const int kh   = lane >> 4;
  const int nl   = lane & 15;
  const int wg   = blockIdx.x;

  // init: zero states, cache small vectors
  for (int i = tid; i < 16 * 128; i += 128) { hS[i] = 0.0f; mS[i] = 0.0f; }
  for (int i = tid; i < 16 * 64; i += 128) {
    int r = i >> 6, j = i & 63;
    int b = wg * 16 + r;
    seS[i] = (_Float16)p.s_emb[p.s_data[b] * 64 + j];
  }
  if (tid < 32) wkS[tid] = p.w_k[tid];
  wo2S[tid] = p.W_o2[tid];
  wprS[tid] = p.W_pref[tid];
  __syncthreads();

  for (int t = 0; t < NSTEP; ++t) {
    // ---- phase A0: indices + correlation softmaxes (1 thread / row) ----
    if (tid < 16) {
      int r = tid, b = wg * 16 + r, base = b * TT;
      int qi = p.q_data[base + t + 1];
      int ai = p.a_data[base + t + 1];
      int li = p.l_data[base + t + 1];
      int di = p.d_data[base + t + 1];
      int di0 = p.d_data[base + t];
      int qn = p.q_data[base + t + 2];
      idxs[r][0] = qi; idxs[r][1] = ai; idxs[r][2] = li;
      idxs[r][3] = di; idxs[r][4] = di0; idxs[r][5] = qn;
      int is = (di == 0);
      isq[r] = is;
      isf[r] = is ? 1.0f : 0.0f;
      const float* lg = is ? (p.q_corr + qi * 32) : (p.l_corr + li * 32);
      float mx = -1e30f;
      for (int c = 0; c < 32; ++c) mx = fmaxf(mx, lg[c]);
      float s = 0.0f;
      for (int c = 0; c < 32; ++c) { float e = __expf(lg[c] - mx); wcur[r * 32 + c] = e; s += e; }
      float inv = rcp_f(s);
      for (int c = 0; c < 32; ++c) wcur[r * 32 + c] *= inv;
      const float* lg2 = p.q_corr + qn * 32;
      mx = -1e30f;
      for (int c = 0; c < 32; ++c) mx = fmaxf(mx, lg2[c]);
      s = 0.0f;
      for (int c = 0; c < 32; ++c) { float e = __expf(lg2[c] - mx); wnxt[r * 32 + c] = e; s += e; }
      inv = rcp_f(s);
      for (int c = 0; c < 32; ++c) wnxt[r * 32 + c] *= inv;
    }
    __syncthreads();

    // ---- phase A1: build GEMM1 A-matrix + next-q embedding ----
    for (int i = tid; i < 16 * KPAD + 16 * 64; i += 128) {
      if (i < 16 * KPAD) {
        int r = i / KPAD, c = i % KPAD;
        int is = isq[r];
        int qi = idxs[r][0], ai = idxs[r][1], li = idxs[r][2];
        int di = idxs[r][3], di0 = idxs[r][4];
        float v;
        if      (c < 64)  v = is ? p.q_emb[qi * 64 + c] : 0.0f;
        else if (c < 128) v = is ? p.a_emb[ai * 64 + (c - 64)] : 0.0f;
        else if (c < 192) v = is ? 0.0f : p.l_emb[li * 64 + (c - 128)];
        else if (c < 320) v = hS[r * 128 + (c - 192)];
        else if (c < 384) v = (float)seS[r * 64 + (c - 320)];
        else if (c < 448) v = is ? p.qb_emb[qi * 64 + (c - 384)] : 0.0f;
        else if (c < 512) v = is ? 0.0f : p.lb_emb[li * 64 + (c - 448)];
        else if (c < 528) v = is ? p.d_emb[di * 16 + (c - 512)] : 0.0f;
        else if (c < 544) v = is ? 0.0f : p.d_emb[di * 16 + (c - 528)];
        else if (c < 560) v = p.d_emb[di * 16 + (c - 544)];
        else              v = p.d_emb[di0 * 16 + (c - 560)];
        uA[i] = (_Float16)v;
      } else {
        int j = i - 16 * KPAD;
        int r = j >> 6, jj = j & 63;
        qnS[j] = (_Float16)p.q_emb[idxs[r][5] * 64 + jj];
      }
    }
    __syncthreads();

    // ---- GEMM1: [16,576] x [576,512] -> erase|add|x|trans ----
    // loop-interchanged: 1 A-fragment load feeds 8 WMMAs; blk == wave.
    {
      v8f z = {};
      v8f acc[8] = {z, z, z, z, z, z, z, z};
      for (int kt = 0; kt < KT1; ++kt) {
        v16h af = load_afrag(uA, kt * 32);
#pragma unroll
        for (int i = 0; i < 8; ++i)
          acc[i] = wmma_f16(af, load_bfrag(W1, kt * NT1 + wave * 8 + i), acc[i]);
      }
#pragma unroll
      for (int i = 0; i < 8; ++i) {
        int nt = wave * 8 + i;
        int n = nt * 16 + nl;
        float bq = bqW[n], bl = blW[n];
#pragma unroll
        for (int r = 0; r < 8; ++r) {
          int m = r + 8 * kh;
          float bias = bl + isf[m] * (bq - bl);
          float v = acc[i][r] + bias;
          if (wave == 0)      v = sigm(v);     // erase
          else if (wave == 1) v = tanh_f(v);   // add
          else if (wave == 3) v = tanh_f(v);   // trans
          outS[m * 512 + n] = (_Float16)v;     // wave 2: x (no activation)
        }
      }
    }
    __syncthreads();

    // ---- phase C: M update + knowledge summary k + read; build GEMM2 A ----
    {
      float bk = p.b_k[0];
      for (int i = tid; i < 16 * 128; i += 128) {
        int r = i >> 7, v = i & 127;
        int b = wg * 16 + r;
        float er  = (float)outS[r * 512 + v];
        float ad  = (float)outS[r * 512 + 128 + v];
        float trn = (float)outS[r * 512 + 384 + v];
        float at = ad + trn;
        float kacc = 0.0f, racc = 0.0f;
        float* Mp = Mws + ((size_t)b * 32) * 128 + v;
        for (int c = 0; c < 32; ++c) {
          float Mv = Mp[c * 128];
          float w = wcur[r * 32 + c];
          Mv = Mv * (1.0f - w * er) + w * at;
          Mp[c * 128] = Mv;
          kacc += wkS[c] * Mv;
          racc += wnxt[r * 32 + c] * Mv;
        }
        float kk = tanh_f(kacc + bk);
        uA[r * KPAD + v]       = outS[r * 512 + 256 + v];    // x
        uA[r * KPAD + 128 + v] = (_Float16)kk;               // k
        uA[r * KPAD + 256 + v] = (_Float16)hS[r * 128 + v];  // h
        if (v < 64) uA[r * KPAD + 384 + v] = seS[r * 64 + v];
        rdS[r * 128 + v] = (_Float16)racc;
      }
    }
    __syncthreads();

    // ---- GEMM2: [16,448] x [448,512] -> i|g|f|o  (gate == wave) ----
    {
      v8f z = {};
      v8f acc[8] = {z, z, z, z, z, z, z, z};
      for (int kt = 0; kt < KT2; ++kt) {
        v16h af = load_afrag(uA, kt * 32);
#pragma unroll
        for (int i = 0; i < 8; ++i)
          acc[i] = wmma_f16(af, load_bfrag(W2, kt * NT2 + wave * 8 + i), acc[i]);
      }
#pragma unroll
      for (int i = 0; i < 8; ++i) {
        int nt = wave * 8 + i;
        int n = nt * 16 + nl;
        float bias = b2W[n];
#pragma unroll
        for (int r = 0; r < 8; ++r) {
          int m = r + 8 * kh;
          float v = acc[i][r] + bias;
          v = (wave == 1) ? tanh_f(v) : sigm(v);
          outS[m * 512 + n] = (_Float16)v;
        }
      }
    }
    __syncthreads();

    // ---- phase E: LSTM state update; build GEMM3 A = [q_next, read, h2] ----
    for (int i = tid; i < 16 * 128; i += 128) {
      int r = i >> 7, v = i & 127;
      float ig = (float)outS[r * 512 + v];
      float gg = (float)outS[r * 512 + 128 + v];
      float fg = (float)outS[r * 512 + 256 + v];
      float og = (float)outS[r * 512 + 384 + v];
      float m2 = fg * mS[i] + ig * gg;
      float h2 = og * tanh_f(m2);
      mS[i] = m2;
      hS[i] = h2;
      uA[r * KPAD + 192 + v] = (_Float16)h2;
      uA[r * KPAD + 64 + v]  = rdS[i];
      if (v < 64) uA[r * KPAD + v] = qnS[r * 64 + v];
    }
    __syncthreads();

    // ---- GEMM3: [16,320] x [320,128] -> summary (tanh) ----
    {
      v8f z = {};
      v8f acc[2] = {z, z};
      for (int kt = 0; kt < KT3; ++kt) {
        v16h af = load_afrag(uA, kt * 32);
#pragma unroll
        for (int i = 0; i < 2; ++i)
          acc[i] = wmma_f16(af, load_bfrag(W3, kt * NT3 + wave * 2 + i), acc[i]);
      }
#pragma unroll
      for (int i = 0; i < 2; ++i) {
        int nt = wave * 2 + i;
        int n = nt * 16 + nl;
        float bias = bo1W[n];
#pragma unroll
        for (int r = 0; r < 8; ++r) {
          int m = r + 8 * kh;
          outS[m * 512 + n] = (_Float16)tanh_f(acc[i][r] + bias);
        }
      }
    }
    __syncthreads();

    // ---- phase G: output heads ----
    if (tid < 16) {
      int r = tid, b = wg * 16 + r;
      float s1 = p.b_o2[0];
      for (int n = 0; n < 128; ++n) s1 += (float)outS[r * 512 + n] * wo2S[n];
      float s2 = p.b_pref[0];
      for (int v = 0; v < 128; ++v) s2 += hS[r * 128 + v] * wprS[v];
      size_t o = ((size_t)b * NSTEP + t) * 2;
      out[o]     = sigm(s1);
      out[o + 1] = sigm(s2);
    }
    __syncthreads();
  }
}

extern "C" void kernel_launch(void* const* d_in, const int* in_sizes, int n_in,
                              void* d_out, int out_size, void* d_ws, size_t ws_size,
                              hipStream_t stream) {
  P p;
  p.q_data = (const int*)d_in[0];
  p.a_data = (const int*)d_in[1];
  p.l_data = (const int*)d_in[2];
  p.d_data = (const int*)d_in[3];
  p.s_data = (const int*)d_in[4];
  p.init_M = (const float*)d_in[5];
  p.q_emb  = (const float*)d_in[6];
  p.l_emb  = (const float*)d_in[7];
  p.a_emb  = (const float*)d_in[8];
  p.d_emb  = (const float*)d_in[9];
  p.s_emb  = (const float*)d_in[10];
  p.qb_emb = (const float*)d_in[11];
  p.lb_emb = (const float*)d_in[12];
  p.q_corr = (const float*)d_in[13];
  p.l_corr = (const float*)d_in[14];
  p.W_EQ = (const float*)d_in[15];  p.b_EQ = (const float*)d_in[16];
  p.W_EL = (const float*)d_in[17];  p.b_EL = (const float*)d_in[18];
  p.W_Ebh = (const float*)d_in[19];
  p.W_Estu = (const float*)d_in[20]; p.b_Estu = (const float*)d_in[21];
  p.W_DQ = (const float*)d_in[22];  p.b_DQ = (const float*)d_in[23];
  p.W_DL = (const float*)d_in[24];  p.b_DL = (const float*)d_in[25];
  p.W_Dbh = (const float*)d_in[26];
  p.W_Dstu = (const float*)d_in[27];
  p.W_tr = (const float*)d_in[28];  p.b_tr = (const float*)d_in[29];
  p.W_o1 = (const float*)d_in[30];  p.b_o1 = (const float*)d_in[31];
  p.W_o2 = (const float*)d_in[32];  p.b_o2 = (const float*)d_in[33];
  p.W_mQ = (const float*)d_in[34];  p.b_mQ = (const float*)d_in[35];
  p.W_mL = (const float*)d_in[36];  p.b_mL = (const float*)d_in[37];
  p.W_pref = (const float*)d_in[38]; p.b_pref = (const float*)d_in[39];
  for (int g = 0; g < 4; ++g) {
    int base = 40 + g * 7;
    p.Wg[g]  = (const float*)d_in[base];
    p.bg[g]  = (const float*)d_in[base + 1];
    p.Wgk[g] = (const float*)d_in[base + 2];
    p.bgk[g] = (const float*)d_in[base + 3];
    p.Wgh[g] = (const float*)d_in[base + 4];
    p.Wgs[g] = (const float*)d_in[base + 5];
    p.bgs[g] = (const float*)d_in[base + 6];
  }
  p.w_k = (const float*)d_in[68];
  p.b_k = (const float*)d_in[69];

  _Float16* W1 = (_Float16*)((char*)d_ws + WS_W1);
  _Float16* W2 = (_Float16*)((char*)d_ws + WS_W2);
  _Float16* W3 = (_Float16*)((char*)d_ws + WS_W3);
  float*    Mw = (float*)((char*)d_ws + WS_M);
  float*    bq = (float*)((char*)d_ws + WS_BQ);
  float*    bl = (float*)((char*)d_ws + WS_BL);
  float*    b2 = (float*)((char*)d_ws + WS_B2);
  float*    bo1 = (float*)((char*)d_ws + WS_BO1);

  const int F1 = KT1 * NT1, F2 = KT2 * NT2, F3 = KT3 * NT3;
  int wtotal = (F1 + F2 + F3) * FRAG_HALVES;
  prep_weights<<<(wtotal + 255) / 256, 256, 0, stream>>>(p, W1, W2, W3);
  prep_M<<<(128 * 32 * 128 + 255) / 256, 256, 0, stream>>>(p.init_M, Mw);
  prep_bias<<<2, 256, 0, stream>>>(p, bq, bl, b2, bo1);
  ktbm_main<<<8, 128, 0, stream>>>(p, W1, W2, W3, Mw, bq, bl, b2, bo1, (float*)d_out);
}
